// MOELayer_86569360818510
// MI455X (gfx1250) — matrix operand order repaired and checked by reading, not compiled, not run on hardware
//
#include <hip/hip_runtime.h>
#include <math.h>

#define BB 4
#define SS 2048
#define DDIM 1024
#define EE 8
#define FF 4096
#define TT (BB * SS)          /* 8192 tokens */
#define CAP (2 * TT / EE)     /* 2048 capacity per expert */

typedef __attribute__((ext_vector_type(16))) __bf16 v16bf;
typedef __attribute__((ext_vector_type(8)))  float  v8f;

static __device__ __forceinline__ unsigned short f32_to_bf16_rn(float f) {
  unsigned u = __float_as_uint(f);
  unsigned r = u + 0x7FFFu + ((u >> 16) & 1u);
  return (unsigned short)(r >> 16);
}

// ---------------------------------------------------------------------------
// fp32 -> bf16 bulk convert (weights), vectorized float4 -> ushort4
// ---------------------------------------------------------------------------
__global__ void cvt_bf16_kernel(const float* __restrict__ src,
                                unsigned short* __restrict__ dst, long n4) {
  long i = (long)blockIdx.x * blockDim.x + threadIdx.x;
  long stride = (long)gridDim.x * blockDim.x;
  for (; i < n4; i += stride) {
    float4 v = ((const float4*)src)[i];
    ushort4 o;
    o.x = f32_to_bf16_rn(v.x);
    o.y = f32_to_bf16_rn(v.y);
    o.z = f32_to_bf16_rn(v.z);
    o.w = f32_to_bf16_rn(v.w);
    ((ushort4*)dst)[i] = o;
  }
}

// ---------------------------------------------------------------------------
// Gating: one wave32 per token. logits = x[t] @ wg, softmax, top1/top2.
// ---------------------------------------------------------------------------
__global__ __launch_bounds__(256) void gate_kernel(
    const float* __restrict__ x, const float* __restrict__ wg,
    int* __restrict__ idx1, int* __restrict__ idx2,
    float* __restrict__ g1o, float* __restrict__ g2o) {
  int wid = threadIdx.x >> 5;
  int lane = threadIdx.x & 31;
  int t = blockIdx.x * 8 + wid;
  const float* xr = x + (size_t)t * DDIM;
  float acc[EE];
#pragma unroll
  for (int e = 0; e < EE; e++) acc[e] = 0.f;
  for (int i = 0; i < DDIM / 32; i++) {
    int d = lane + i * 32;
    float xv = xr[d];
    const float4* wr = (const float4*)(wg + (size_t)d * EE);
    float4 w0 = wr[0], w1 = wr[1];
    acc[0] += xv * w0.x; acc[1] += xv * w0.y;
    acc[2] += xv * w0.z; acc[3] += xv * w0.w;
    acc[4] += xv * w1.x; acc[5] += xv * w1.y;
    acc[6] += xv * w1.z; acc[7] += xv * w1.w;
  }
#pragma unroll
  for (int e = 0; e < EE; e++)
    for (int off = 16; off; off >>= 1) acc[e] += __shfl_xor(acc[e], off, 32);
  if (lane == 0) {
    float m = acc[0];
    for (int e = 1; e < EE; e++) m = fmaxf(m, acc[e]);
    float ex[EE], s = 0.f;
    for (int e = 0; e < EE; e++) { ex[e] = expf(acc[e] - m); s += ex[e]; }
    // top1 = first strict max (matches argmax tie rule)
    int i1 = 0;
    for (int e = 1; e < EE; e++) if (acc[e] > acc[i1]) i1 = e;
    int i2 = (i1 == 0) ? 1 : 0;
    for (int e = 0; e < EE; e++) {
      if (e == i1) continue;
      if (acc[e] > acc[i2]) i2 = e;
    }
    float g1 = ex[i1] / s, g2 = ex[i2] / s;
    float denom = fmaxf(g1 + g2, 1e-9f);
    idx1[t] = i1; idx2[t] = i2;
    g1o[t] = g1 / denom; g2o[t] = g2 / denom;
  }
}

// ---------------------------------------------------------------------------
// Ordered capacity scan (single 1024-thread block, 8 tokens per thread).
// Exact GShard semantics: loc1 = exclusive cumsum(mask1); loc2 offset by
// total count1[e]. Produces slot ids, keep-masked weights, slot->token map.
// ---------------------------------------------------------------------------
__global__ __launch_bounds__(1024) void scan_kernel(
    const int* __restrict__ idx1, const int* __restrict__ idx2,
    const float* __restrict__ g1, const float* __restrict__ g2,
    int* __restrict__ slot1, int* __restrict__ slot2,
    float* __restrict__ w1, float* __restrict__ w2,
    int* __restrict__ slot_token) {
  __shared__ int wsum[32];
  __shared__ int sexcl[32];
  __shared__ int stot;
  __shared__ int tot1[EE];
  const int tid = threadIdx.x;
  const int lane = tid & 31;
  const int wid = tid >> 5;
  const int TP = TT / 1024;  // 8 tokens per thread
  const int base = tid * TP;

  for (int i = tid; i < EE * CAP; i += 1024) slot_token[i] = -1;

  auto exscan = [&](int v, int& total) -> int {
    int incl = v;
    for (int off = 1; off < 32; off <<= 1) {
      int n = __shfl_up(incl, off, 32);
      if (lane >= off) incl += n;
    }
    if (lane == 31) wsum[wid] = incl;
    __syncthreads();
    if (wid == 0) {
      int wv = wsum[lane];
      int winc = wv;
      for (int off = 1; off < 32; off <<= 1) {
        int n = __shfl_up(winc, off, 32);
        if (lane >= off) winc += n;
      }
      sexcl[lane] = winc - wv;
      if (lane == 31) stot = winc;
    }
    __syncthreads();
    total = stot;
    int r = sexcl[wid] + (incl - v);
    __syncthreads();
    return r;
  };

  // ---- first choice ----
  int cnt[EE];
#pragma unroll
  for (int e = 0; e < EE; e++) cnt[e] = 0;
  for (int j = 0; j < TP; j++) cnt[idx1[base + j]]++;
  int b1[EE];
  for (int e = 0; e < EE; e++) {
    int tot;
    b1[e] = exscan(cnt[e], tot);
    if (tid == 0) tot1[e] = tot;
  }
  {
    int run[EE];
#pragma unroll
    for (int e = 0; e < EE; e++) run[e] = b1[e];
    for (int j = 0; j < TP; j++) {
      int t = base + j;
      int e = idx1[t];
      int loc = run[e]++;
      int keep = loc < CAP;
      int slot = e * CAP + (loc < CAP ? loc : CAP - 1);
      slot1[t] = slot;
      w1[t] = keep ? g1[t] : 0.f;
      if (keep) slot_token[slot] = t;
    }
  }
  // ---- second choice (offset by total count1[e]) ----
#pragma unroll
  for (int e = 0; e < EE; e++) cnt[e] = 0;
  for (int j = 0; j < TP; j++) cnt[idx2[base + j]]++;
  int b2[EE];
  for (int e = 0; e < EE; e++) {
    int tot;
    b2[e] = exscan(cnt[e], tot);
  }
  {
    int run[EE];
#pragma unroll
    for (int e = 0; e < EE; e++) run[e] = tot1[e] + b2[e];
    for (int j = 0; j < TP; j++) {
      int t = base + j;
      int e = idx2[t];
      int loc = run[e]++;
      int keep = loc < CAP;
      int slot = e * CAP + (loc < CAP ? loc : CAP - 1);
      slot2[t] = slot;
      w2[t] = keep ? g2[t] : 0.f;
      if (keep) slot_token[slot] = t;
    }
  }
}

// ---------------------------------------------------------------------------
// Dispatch gather: disp[slot] = bf16(x[slot_token[slot]]) or 0.
// ---------------------------------------------------------------------------
__global__ __launch_bounds__(128) void dispatch_kernel(
    const float* __restrict__ x, const int* __restrict__ slot_token,
    unsigned short* __restrict__ disp) {
  int s = blockIdx.x;
  int t = slot_token[s];
  ushort4* dst = (ushort4*)(disp + (size_t)s * DDIM);
  if (t >= 0) {
    const float4* src = (const float4*)(x + (size_t)t * DDIM);
    for (int i = threadIdx.x; i < DDIM / 4; i += 128) {
      float4 v = src[i];
      ushort4 o;
      o.x = f32_to_bf16_rn(v.x);
      o.y = f32_to_bf16_rn(v.y);
      o.z = f32_to_bf16_rn(v.z);
      o.w = f32_to_bf16_rn(v.w);
      dst[i] = o;
    }
  } else {
    ushort4 z; z.x = 0; z.y = 0; z.z = 0; z.w = 0;
    for (int i = threadIdx.x; i < DDIM / 4; i += 128) dst[i] = z;
  }
}

// ---------------------------------------------------------------------------
// WMMA GEMM: C[M,N] = A[M,K](bf16,row) * B[K,N](bf16,row), fp32 accum.
// Block tile 128x256, 8 waves of 64x64, BK=64 (two wmma K-substeps/stage,
// 32 WMMAs per barrier pair). Register-staged software pipeline: next tile's
// global_load_b128s are issued into VGPRs while the current tile's WMMAs run
// from LDS, so HBM/L2 latency overlaps the matrix unit.
// GELU=true: fused exact GELU, bf16 output. GELU=false: fp32 output.
// Fragment layouts follow CDNA5 ISA 7.12.2 (16-bit A 16x32, B 32x16).
// ---------------------------------------------------------------------------
#define BM 128
#define BN 256
#define BK 64
#define LDS_S 66  /* padded row stride (bf16 units) to dodge bank conflicts */

template <bool GELU>
__global__ __launch_bounds__(256) void gemm_kernel(
    const unsigned short* __restrict__ A, int lda,
    const unsigned short* __restrict__ B, int ldb,
    void* __restrict__ Cout, int ldc, int K) {
  __shared__ unsigned short ldsA[BM * LDS_S];  // [128][64+pad]
  __shared__ unsigned short ldsB[BN * LDS_S];  // transposed [N=256][64+pad]

  const int tid = threadIdx.x;
  const int lane = tid & 31;
  const int waveId = tid >> 5;
  const int waveM = waveId & 1;   // 2 waves over M
  const int waveN = waveId >> 1;  // 4 waves over N
  const int l16 = lane & 15;
  const int lhi = lane >> 4;      // 0/1: selects K-half per ISA layout

  const int blockM = blockIdx.y * BM;
  const int blockN = blockIdx.x * BN;

  v8f acc[4][4];
#pragma unroll
  for (int m = 0; m < 4; m++)
#pragma unroll
    for (int n = 0; n < 4; n++)
#pragma unroll
      for (int r = 0; r < 8; r++) acc[m][n][r] = 0.f;

  union FragAB { v16bf v; unsigned int u[8]; };

  uint4 regA[4];  // A tile 128x64: 1024 chunks of 8 bf16 -> 4/thread
  uint4 regB[8];  // B tile 64x256: 2048 chunks of 8 bf16 -> 8/thread

  auto loadRegs = [&](int kt) {
#pragma unroll
    for (int i = 0; i < 4; i++) {
      int cid = tid + i * 256;
      int row = cid >> 3;             // 8 chunks per 64-wide row
      int col = (cid & 7) << 3;
      regA[i] = *(const uint4*)(A + (size_t)(blockM + row) * lda + kt + col);
    }
#pragma unroll
    for (int i = 0; i < 8; i++) {
      int cid = tid + i * 256;
      int krow = cid >> 5;            // 32 chunks per 256-wide row
      int col = (cid & 31) << 3;
      regB[i] = *(const uint4*)(B + (size_t)(kt + krow) * ldb + blockN + col);
    }
  };

  auto stageToLds = [&]() {
#pragma unroll
    for (int i = 0; i < 4; i++) {
      int cid = tid + i * 256;
      int row = cid >> 3;
      int col = (cid & 7) << 3;
      unsigned short* d = &ldsA[row * LDS_S + col];
      *(unsigned int*)(d + 0) = regA[i].x;
      *(unsigned int*)(d + 2) = regA[i].y;
      *(unsigned int*)(d + 4) = regA[i].z;
      *(unsigned int*)(d + 6) = regA[i].w;
    }
#pragma unroll
    for (int i = 0; i < 8; i++) {
      int cid = tid + i * 256;
      int krow = cid >> 5;
      int col = (cid & 31) << 3;
      union { uint4 u; unsigned short s[8]; } cv;
      cv.u = regB[i];
#pragma unroll
      for (int j = 0; j < 8; j++) ldsB[(col + j) * LDS_S + krow] = cv.s[j];
    }
  };

  auto computeTile = [&]() {
#pragma unroll
    for (int s = 0; s < 2; s++) {  // two 16x16x32 K-substeps per stage
      FragAB aF[4], bF[4];
#pragma unroll
      for (int m = 0; m < 4; m++) {
        const unsigned short* baseA =
            &ldsA[(waveM * 64 + m * 16 + l16) * LDS_S + s * 32];
#pragma unroll
        for (int r = 0; r < 8; r++) {
          int kb = ((r < 4) ? (2 * r) : (2 * r + 8)) + (lhi << 3);
          aF[m].u[r] = *(const unsigned int*)(baseA + kb);
        }
      }
#pragma unroll
      for (int n = 0; n < 4; n++) {
        const unsigned short* baseB =
            &ldsB[(waveN * 64 + n * 16 + l16) * LDS_S + s * 32];
#pragma unroll
        for (int r = 0; r < 8; r++) {
          int kb = ((r < 4) ? (2 * r) : (2 * r + 8)) + (lhi << 3);
          bF[n].u[r] = *(const unsigned int*)(baseB + kb);
        }
      }
#pragma unroll
      for (int m = 0; m < 4; m++)
#pragma unroll
        for (int n = 0; n < 4; n++)
          acc[m][n] = __builtin_amdgcn_wmma_f32_16x16x32_bf16(
              false, aF[m].v, false, bF[n].v, (short)0, acc[m][n], false,
              false);
    }
  };

  // ---- software pipeline: prologue fills LDS with tile 0 ----
  loadRegs(0);
  stageToLds();
  __syncthreads();

  for (int kt = BK; kt < K; kt += BK) {
    loadRegs(kt);     // global loads for next tile fly during compute
    computeTile();    // 32 WMMAs from LDS
    __syncthreads();
    stageToLds();     // drain regs -> LDS (waits on loadcnt here)
    __syncthreads();
  }
  computeTile();      // last tile

  // ---- epilogue: C/D layout lanes0-15 N=lane,M=r; lanes16-31 N=lane-16,M=r+8
#pragma unroll
  for (int m = 0; m < 4; m++) {
#pragma unroll
    for (int n = 0; n < 4; n++) {
#pragma unroll
      for (int r = 0; r < 8; r++) {
        int row = blockM + waveM * 64 + m * 16 + r + (lhi << 3);
        int col = blockN + waveN * 64 + n * 16 + l16;
        float v = acc[m][n][r];
        if constexpr (GELU) {
          v = 0.5f * v * (1.f + erff(v * 0.70710678118654752f));
          ((unsigned short*)Cout)[(size_t)row * ldc + col] = f32_to_bf16_rn(v);
        } else {
          ((float*)Cout)[(size_t)row * ldc + col] = v;
        }
      }
    }
  }
}

// ---------------------------------------------------------------------------
// Combine: out[t] = w1[t]*eo[f1[t]] + w2[t]*eo[f2[t]]
// ---------------------------------------------------------------------------
__global__ __launch_bounds__(256) void combine_kernel(
    const float* __restrict__ eo, const int* __restrict__ slot1,
    const int* __restrict__ slot2, const float* __restrict__ w1,
    const float* __restrict__ w2, float* __restrict__ out) {
  int t = blockIdx.x;
  int f1 = slot1[t], f2 = slot2[t];
  float a = w1[t], b = w2[t];
  const float4* e1 = (const float4*)(eo + (size_t)f1 * DDIM);
  const float4* e2 = (const float4*)(eo + (size_t)f2 * DDIM);
  float4* o = (float4*)(out + (size_t)t * DDIM);
  for (int i = threadIdx.x; i < DDIM / 4; i += 256) {
    float4 v1 = e1[i], v2 = e2[i];
    float4 r;
    r.x = a * v1.x + b * v2.x;
    r.y = a * v1.y + b * v2.y;
    r.z = a * v1.z + b * v2.z;
    r.w = a * v1.w + b * v2.w;
    o[i] = r;
  }
}

// ---------------------------------------------------------------------------
extern "C" void kernel_launch(void* const* d_in, const int* in_sizes, int n_in,
                              void* d_out, int out_size, void* d_ws,
                              size_t ws_size, hipStream_t stream) {
  (void)in_sizes; (void)n_in; (void)out_size; (void)ws_size;
  const float* x      = (const float*)d_in[0];  // [B,S,D]
  const float* wg     = (const float*)d_in[1];  // [D,E]
  const float* w_gate = (const float*)d_in[2];  // [E,D,F]
  const float* w_down = (const float*)d_in[3];  // [E,F,D]
  float* out = (float*)d_out;

  char* ws = (char*)d_ws;
  size_t off = 0;
  auto alloc = [&](size_t bytes) -> char* {
    char* p = ws + off;
    off = (off + bytes + 255) & ~(size_t)255;
    return p;
  };

  int* idx1 = (int*)alloc(sizeof(int) * TT);
  int* idx2 = (int*)alloc(sizeof(int) * TT);
  float* g1 = (float*)alloc(sizeof(float) * TT);
  float* g2 = (float*)alloc(sizeof(float) * TT);
  float* w1 = (float*)alloc(sizeof(float) * TT);
  float* w2 = (float*)alloc(sizeof(float) * TT);
  int* slot1 = (int*)alloc(sizeof(int) * TT);
  int* slot2 = (int*)alloc(sizeof(int) * TT);
  int* slot_token = (int*)alloc(sizeof(int) * EE * CAP);
  unsigned short* disp = (unsigned short*)alloc(2ull * EE * CAP * DDIM);
  unsigned short* wgbf = (unsigned short*)alloc(2ull * EE * DDIM * FF);
  unsigned short* wdbf = (unsigned short*)alloc(2ull * EE * FF * DDIM);
  unsigned short* hbf  = (unsigned short*)alloc(2ull * CAP * FF);  // per-expert
  float* eo = (float*)alloc(4ull * EE * CAP * DDIM);

  long n4w = (long)EE * DDIM * FF / 4;
  cvt_bf16_kernel<<<4096, 256, 0, stream>>>(w_gate, wgbf, n4w);
  cvt_bf16_kernel<<<4096, 256, 0, stream>>>(w_down, wdbf, n4w);

  gate_kernel<<<TT / 8, 256, 0, stream>>>(x, wg, idx1, idx2, g1, g2);
  scan_kernel<<<1, 1024, 0, stream>>>(idx1, idx2, g1, g2, slot1, slot2, w1, w2,
                                      slot_token);
  dispatch_kernel<<<EE * CAP, 128, 0, stream>>>(x, slot_token, disp);

  for (int e = 0; e < EE; e++) {
    const unsigned short* Ae = disp + (size_t)e * CAP * DDIM;
    const unsigned short* Bg = wgbf + (size_t)e * DDIM * FF;
    gemm_kernel<true><<<dim3(FF / BN, CAP / BM), 256, 0, stream>>>(
        Ae, DDIM, Bg, FF, hbf, FF, DDIM);
    const unsigned short* Bd = wdbf + (size_t)e * FF * DDIM;
    gemm_kernel<false><<<dim3(DDIM / BN, CAP / BM), 256, 0, stream>>>(
        hbf, FF, Bd, DDIM, eo + (size_t)e * CAP * DDIM, DDIM, FF);
  }

  combine_kernel<<<TT, 256, 0, stream>>>(eo, slot1, slot2, w1, w2, out);
}
